// HistoryEncoder_18090402250870
// MI455X (gfx1250) — matrix operand order repaired and checked by reading, not compiled, not run on hardware
//
#include <hip/hip_runtime.h>
#include <cmath>

// ---------------- problem constants ----------------
#define BSZ    4
#define SEQL   4096
#define DM     512
#define DI     1024     // d_inner
#define NH     16       // heads
#define HD     64       // headdim
#define DSTATE 64
#define DCONV  4
#define CHUNK  256
#define NC     16       // SEQL / CHUNK
#define CONVD  1152     // DI + 2*DSTATE
#define DPROJ  2192     // 2*DI + 2*DSTATE + NH
#define MTOT   16384    // BSZ*SEQL

typedef _Float16 h16;
typedef __attribute__((ext_vector_type(16))) _Float16 v16h;
typedef __attribute__((ext_vector_type(8)))  float    v8f;

union FragU { v16h v; int4 q[2]; h16 h[16]; };

__device__ inline v8f vzero8() { v8f z; for (int i = 0; i < 8; ++i) z[i] = 0.f; return z; }

// A-operand fragment (16x32 f16): lane&15 = M row (caller bakes row into p).
// halves 0..7  = K = grp*8 .. grp*8+7 ; halves 8..15 = K = grp*8+16 .. grp*8+23
__device__ inline v16h loadA(const h16* p, int grp) {
  FragU f;
  const h16* q = p + grp * 8;
  f.q[0] = *(const int4*)(q);
  f.q[1] = *(const int4*)(q + 16);
  return f.v;
}

// B-operand fragment (32x16 f16): lane&15 = N col (caller bakes col into p).
// halves 0..7  = K = grp*8 .. grp*8+7 ; halves 8..15 = K = grp*8+16 .. grp*8+23
__device__ inline v16h loadB(const h16* p, int grp) {
  FragU f;
  const h16* q = p + grp * 8;
  f.q[0] = *(const int4*)(q);
  f.q[1] = *(const int4*)(q + 16);
  return f.v;
}

__device__ inline v8f wmma16(v16h a, v16h b, v8f c) {
  return __builtin_amdgcn_wmma_f32_16x16x32_f16(false, a, false, b, (short)0, c, false, false);
}

__device__ inline float silu(float x) { return x / (1.f + __expf(-x)); }

// async global -> LDS, 16B per lane, tracked by ASYNCcnt.
// LDS byte offset = low 32 bits of the generic pointer (ISA 10.2: LDS aperture truncation).
__device__ inline void async_b128_to_lds(void* ldsDst, const void* g) {
  unsigned dst = (unsigned)(uintptr_t)ldsDst;
  asm volatile("global_load_async_to_lds_b128 %0, %1, off"
               :: "v"(dst), "v"(g) : "memory");
}

// ---------------- fp32 -> f16 convert ----------------
__global__ void cvt_f32_f16(const float* __restrict__ src, h16* __restrict__ dst, int n) {
  int i = blockIdx.x * blockDim.x + threadIdx.x;
  if (i < n) dst[i] = (h16)src[i];
}

// ---------------- NT GEMM: C[M][N] = A16[M][K] * Bt16[N][K]^T ----------------
// block: 256 threads (8 waves). tile: 128(M) x 128(N). K multiple of 32.
// B tile double-buffered in LDS via async DMA; A fragments register-prefetched.
// Last K-step peeled so the hot loop is branch-free (single WMMA block, no acc copies).
template <typename OutT>
__global__ __launch_bounds__(256) void gemm_nt_f16(
    const h16* __restrict__ A, const h16* __restrict__ Bt, OutT* __restrict__ C,
    int M, int N, int K, int ldc) {
  const int tid   = threadIdx.x;
  const int lane  = tid & 31;
  const int wave  = tid >> 5;
  const int grp   = lane >> 4;
  const int lane15 = lane & 15;
  const int m0    = blockIdx.x * 128 + wave * 16;
  const int n0    = blockIdx.y * 128;

  __shared__ h16 Bs[2][128][32];    // 2 x 8 KB double buffer

  // staging map: 512 x 16B segments per stage, 2 per thread
  const int r0 = tid >> 1;                  // 0..127
  const int g0 = (tid & 1) * 2;             // segment 0 or 2
  int nr0 = n0 + r0; if (nr0 > N - 1) nr0 = N - 1;
  const h16* gB0 = Bt + (size_t)nr0 * K + g0 * 8;

  const h16* Arow = A + (size_t)(m0 + lane15) * K;

  // stage 0: two consecutive 16B segments per thread
  async_b128_to_lds(&Bs[0][r0][g0 * 8], gB0);
  async_b128_to_lds(&Bs[0][r0][(g0 + 1) * 8], gB0 + 8);

  v8f acc[8] = {vzero8(), vzero8(), vzero8(), vzero8(),
                vzero8(), vzero8(), vzero8(), vzero8()};
  v16h aCur = loadA(Arow, grp);

  int buf = 0;
  int k = 0;
  for (; k < K - 32; k += 32) {
    // issue next stage + A prefetch, then wait only for current stage (<=2 outstanding)
    async_b128_to_lds(&Bs[buf ^ 1][r0][g0 * 8], gB0 + k + 32);
    async_b128_to_lds(&Bs[buf ^ 1][r0][(g0 + 1) * 8], gB0 + k + 40);
    v16h aNext = loadA(Arow + k + 32, grp);
    asm volatile("s_wait_asynccnt 0x2" ::: "memory");
    __syncthreads();                       // Bs[buf] visible to all waves
    for (int t = 0; t < 8; ++t) {
      v16h b = loadB(&Bs[buf][t * 16 + lane15][0], grp);
      acc[t] = wmma16(aCur, b, acc[t]);
    }
    __syncthreads();                       // all reads done before buffer reuse
    aCur = aNext;
    buf ^= 1;
  }
  asm volatile("s_wait_asynccnt 0x0" ::: "memory");
  __syncthreads();
  for (int t = 0; t < 8; ++t) {
    v16h b = loadB(&Bs[buf][t * 16 + lane15][0], grp);
    acc[t] = wmma16(aCur, b, acc[t]);
  }

  for (int t = 0; t < 8; ++t) {
    int n = n0 + t * 16 + lane15;
    if (n >= N) continue;
    for (int r = 0; r < 8; ++r) {
      int m = m0 + r + grp * 8;
      C[(size_t)m * ldc + n] = (OutT)acc[t][r];
    }
  }
}

// ---------------- dt = softplus(raw + bias); a = dt * (-exp(A_log)) ----------------
__global__ void dt_kernel(const h16* __restrict__ zx, const float* __restrict__ dt_bias,
                          const float* __restrict__ A_log,
                          float* __restrict__ dtBuf, float* __restrict__ aBuf) {
  int i = blockIdx.x * blockDim.x + threadIdx.x;  // over MTOT*NH
  if (i >= MTOT * NH) return;
  int h = i & (NH - 1), m = i >> 4;
  float x = (float)zx[(size_t)m * DPROJ + (DPROJ - NH) + h] + dt_bias[h];
  float sp = (x > 20.f) ? x : log1pf(__expf(x));
  dtBuf[i] = sp;
  aBuf[i]  = -__expf(A_log[h]) * sp;
}

// ---------------- causal conv1d(4) + SiLU; also xdt = x * dt ----------------
__global__ void conv_kernel(const h16* __restrict__ zx, const float* __restrict__ conv_w,
                            const float* __restrict__ conv_b, const float* __restrict__ dtBuf,
                            h16* __restrict__ xBCc, h16* __restrict__ xdt) {
  long i = (long)blockIdx.x * blockDim.x + threadIdx.x;  // over MTOT*CONVD
  if (i >= (long)MTOT * CONVD) return;
  int c = (int)(i % CONVD);
  int m = (int)(i / CONVD);
  int l = m & (SEQL - 1);
  float acc = conv_b[c];
  for (int k = 0; k < DCONV; ++k) {
    int l2 = l + k - 3;
    if (l2 >= 0) acc += (float)zx[(size_t)(m + k - 3) * DPROJ + DI + c] * conv_w[c * DCONV + k];
  }
  float s = silu(acc);
  xBCc[(size_t)m * CONVD + c] = (h16)s;
  if (c < DI) xdt[(size_t)m * DI + c] = (h16)(s * dtBuf[m * NH + (c >> 6)]);
}

// ---------------- per-chunk inclusive cumsum of a; chunk sums ----------------
__global__ __launch_bounds__(256) void cumsum_kernel(const float* __restrict__ aBuf,
                                                     float* __restrict__ acum,
                                                     float* __restrict__ csum) {
  int gid = blockIdx.x;           // (b*NH+h)*NC + c
  int t = threadIdx.x;
  int c = gid & (NC - 1);
  int bh = gid >> 4;              // b*NH + h
  int h = bh & (NH - 1);
  int b = bh >> 4;
  int m = b * SEQL + c * CHUNK + t;
  __shared__ float s[CHUNK];
  s[t] = aBuf[m * NH + h];
  __syncthreads();
  for (int off = 1; off < CHUNK; off <<= 1) {
    float x = (t >= off) ? s[t - off] : 0.f;
    __syncthreads();
    s[t] += x;
    __syncthreads();
  }
  acum[(size_t)gid * CHUNK + t] = s[t];
  if (t == CHUNK - 1) csum[gid] = s[t];
}

// ---------------- per-chunk states: S[p][n] = sum_l x*dt*decay[l][p] * B[l][n] ----------------
__global__ __launch_bounds__(256) void chunk_states_kernel(
    const h16* __restrict__ xdt, const h16* __restrict__ xBCc,
    const float* __restrict__ acum, float* __restrict__ Sc) {
  const int gid = blockIdx.x;           // b*NC*NH + c*NH + h
  const int h = gid & (NH - 1);
  const int c = (gid >> 4) & (NC - 1);
  const int b = gid >> 8;
  const int t = threadIdx.x;
  const int lane = t & 31, wave = t >> 5, grp = lane >> 4;

  __shared__ h16 xdT[HD][CHUNK];    // 32 KB : decay-scaled x*dt transposed
  __shared__ h16 bT[DSTATE][CHUNK]; // 32 KB : B transposed
  const float* acr = acum + (size_t)((b * NH + h) * NC + c) * CHUNK;
  const float aLast = acr[CHUNK - 1];

  // vectorized transpose staging: 16B global load, 8x b16 LDS scatter
  for (int s = t; s < CHUNK * 8; s += 256) {
    int l = s >> 3, p0 = (s & 7) * 8;
    int m = b * SEQL + c * CHUNK + l;
    float decay = __expf(aLast - acr[l]);
    FragU u; u.q[0] = *(const int4*)(xdt + (size_t)m * DI + h * HD + p0);
    for (int j = 0; j < 8; ++j) xdT[p0 + j][l] = (h16)((float)u.h[j] * decay);
    FragU v; v.q[0] = *(const int4*)(xBCc + (size_t)m * CONVD + DI + p0);
    for (int j = 0; j < 8; ++j) bT[p0 + j][l] = v.h[j];
  }
  __syncthreads();

  for (int j = 0; j < 2; ++j) {
    int tt = wave * 2 + j;          // 16 tiles of 16x16
    int pt = tt >> 2, nt = tt & 3;
    v8f acc = vzero8();
    for (int k = 0; k < CHUNK; k += 32) {
      v16h a = loadA(&xdT[pt * 16 + (lane & 15)][k], grp);
      v16h bfr = loadB(&bT[nt * 16 + (lane & 15)][k], grp);
      acc = wmma16(a, bfr, acc);
    }
    for (int r = 0; r < 8; ++r) {
      int p = pt * 16 + r + grp * 8;
      int n = nt * 16 + (lane & 15);
      Sc[((size_t)gid * HD + p) * DSTATE + n] = acc[r];
    }
  }
}

// ---------------- 16-chunk sequential state scan ----------------
__global__ void state_scan_kernel(const float* __restrict__ Sc, const float* __restrict__ csum,
                                  h16* __restrict__ Sin16) {
  int i = blockIdx.x * blockDim.x + threadIdx.x;  // over BSZ*NH*HD*DSTATE
  if (i >= BSZ * NH * HD * DSTATE) return;
  int n = i & 63, p = (i >> 6) & 63, h = (i >> 12) & 15, b = i >> 16;
  float R = 0.f;
  for (int c = 0; c < NC; ++c) {
    size_t idx = ((((size_t)(b * NC + c) * NH + h) * HD + p) * DSTATE + n);
    Sin16[idx] = (h16)R;
    R = __expf(csum[(b * NH + h) * NC + c]) * R + Sc[idx];
  }
}

// ---------------- chunk Y: Y = (mask(C B^T) * L) @ xdt + exp(acum)*C @ Sin + D*x ----------------
__global__ __launch_bounds__(256) void chunk_y_kernel(
    const h16* __restrict__ xBCc, const h16* __restrict__ xdt, const h16* __restrict__ Sin16,
    const float* __restrict__ acum, const float* __restrict__ Dp, float* __restrict__ yBuf) {
  const int gid = blockIdx.x;
  const int h = gid & (NH - 1);
  const int c = (gid >> 4) & (NC - 1);
  const int b = gid >> 8;
  const int t = threadIdx.x;
  const int lane = t & 31, wave = t >> 5, grp = lane >> 4;

  __shared__ h16 xdT2[HD][CHUNK + 8];    // ~33 KB, x*dt transposed (no decay)
  __shared__ float acs[CHUNK];           // 1 KB
  __shared__ h16 cbT[8][16][32];         // 8 KB per-wave CB staging

  const float* acr = acum + (size_t)((b * NH + h) * NC + c) * CHUNK;
  for (int i = t; i < CHUNK; i += 256) acs[i] = acr[i];
  for (int s = t; s < CHUNK * 8; s += 256) {
    int l = s >> 3, p0 = (s & 7) * 8;
    int m = b * SEQL + c * CHUNK + l;
    FragU u; u.q[0] = *(const int4*)(xdt + (size_t)m * DI + h * HD + p0);
    for (int j = 0; j < 8; ++j) xdT2[p0 + j][l] = u.h[j];
  }
  __syncthreads();

  const float Dh = Dp[h];
  for (int j = 0; j < 2; ++j) {
    const int lt = wave + j * 8;             // l-tile 0..15
    const int m0 = b * SEQL + c * CHUNK + lt * 16;
    const h16* Crow = xBCc + (size_t)(m0 + (lane & 15)) * CONVD + DI + DSTATE;

    v8f accY[4] = {vzero8(), vzero8(), vzero8(), vzero8()};

    // ---- Y_diag over 32-wide s blocks (causal) ----
    const int nsb = lt / 2 + 1;
    for (int sb = 0; sb < nsb; ++sb) {
      for (int half = 0; half < 2; ++half) {
        int s0 = sb * 32 + half * 16;
        const h16* Brow = xBCc + (size_t)(b * SEQL + c * CHUNK + s0 + (lane & 15)) * CONVD + DI;
        v8f cb = vzero8();
        for (int k = 0; k < DSTATE; k += 32) {
          v16h aC = loadA(Crow + k, grp);
          v16h bB = loadB(Brow + k, grp);
          cb = wmma16(aC, bB, cb);
        }
        for (int r = 0; r < 8; ++r) {
          int l = lt * 16 + r + grp * 8;
          int s = s0 + (lane & 15);
          float f = (l >= s) ? __expf(acs[l] - acs[s]) : 0.f;
          cbT[wave][r + grp * 8][half * 16 + (lane & 15)] = (h16)(cb[r] * f);
        }
      }
      asm volatile("s_wait_dscnt 0" ::: "memory");
      v16h aCB = loadA(&cbT[wave][lane & 15][0], grp);
      for (int pt = 0; pt < 4; ++pt) {
        v16h bX = loadB(&xdT2[pt * 16 + (lane & 15)][sb * 32], grp);
        accY[pt] = wmma16(aCB, bX, accY[pt]);
      }
      asm volatile("s_wait_dscnt 0" ::: "memory");
    }

    // ---- Y_off: exp(acum[l]) * C @ Sin ----
    const float eAl = __expf(acs[lt * 16 + (lane & 15)]);
    for (int k = 0; k < DSTATE; k += 32) {
      FragU u; u.v = loadA(Crow + k, grp);
      for (int i = 0; i < 16; ++i) u.h[i] = (h16)((float)u.h[i] * eAl);
      for (int pt = 0; pt < 4; ++pt) {
        const h16* Srow = Sin16 + (((size_t)gid * HD) + pt * 16 + (lane & 15)) * DSTATE + k;
        v16h bS = loadB(Srow, grp);
        accY[pt] = wmma16(u.v, bS, accY[pt]);
      }
    }

    // ---- epilogue: + D*x, store fp32 ----
    for (int pt = 0; pt < 4; ++pt) {
      for (int r = 0; r < 8; ++r) {
        int m = m0 + r + grp * 8;
        int col = h * HD + pt * 16 + (lane & 15);
        float xh = (float)xBCc[(size_t)m * CONVD + col];
        yBuf[(size_t)m * DI + col] = accY[pt][r] + Dh * xh;
      }
    }
  }
}

// ---------------- gate with SiLU(z), RMSnorm, -> f16 ----------------
__global__ __launch_bounds__(256) void gate_norm_kernel(
    const float* __restrict__ yBuf, const h16* __restrict__ zx,
    const float* __restrict__ norm_w, h16* __restrict__ yn) {
  const int m = blockIdx.x;
  const int t = threadIdx.x;
  float g[4];
  float ss = 0.f;
  for (int j = 0; j < 4; ++j) {
    int i = t + j * 256;
    float y = yBuf[(size_t)m * DI + i];
    float z = (float)zx[(size_t)m * DPROJ + i];
    g[j] = y * silu(z);
    ss += g[j] * g[j];
  }
  __shared__ float red[256];
  red[t] = ss;
  __syncthreads();
  for (int off = 128; off > 0; off >>= 1) {
    if (t < off) red[t] += red[t + off];
    __syncthreads();
  }
  float r = rsqrtf(red[0] / (float)DI + 1e-5f);
  for (int j = 0; j < 4; ++j) {
    int i = t + j * 256;
    yn[(size_t)m * DI + i] = (h16)(g[j] * r * norm_w[i]);
  }
}

// ---------------- workspace layout (bytes) ----------------
static const size_t OFF_ZX   = 0;                                  // f16 now
static const size_t OFF_X16  = OFF_ZX   + (size_t)MTOT * DPROJ * 2;
static const size_t OFF_W16  = OFF_X16  + (size_t)MTOT * DM * 2;
static const size_t OFF_WO16 = OFF_W16  + (size_t)DPROJ * DM * 2;
static const size_t OFF_XBC  = OFF_WO16 + (size_t)DM * DI * 2;
static const size_t OFF_XDT  = OFF_XBC  + (size_t)MTOT * CONVD * 2;
static const size_t OFF_DT   = OFF_XDT  + (size_t)MTOT * DI * 2;
static const size_t OFF_A    = OFF_DT   + (size_t)MTOT * NH * 4;
static const size_t OFF_ACUM = OFF_A    + (size_t)MTOT * NH * 4;
static const size_t OFF_CSUM = OFF_ACUM + (size_t)MTOT * NH * 4;
static const size_t OFF_SC   = OFF_CSUM + 4096;
static const size_t OFF_SIN  = OFF_SC   + (size_t)BSZ * NC * NH * HD * DSTATE * 4;
static const size_t OFF_Y    = OFF_SIN  + (size_t)BSZ * NC * NH * HD * DSTATE * 2;
static const size_t OFF_YN   = OFF_Y    + (size_t)MTOT * DI * 4;

extern "C" void kernel_launch(void* const* d_in, const int* in_sizes, int n_in,
                              void* d_out, int out_size, void* d_ws, size_t ws_size,
                              hipStream_t stream) {
  const float* x_seq     = (const float*)d_in[0];
  const float* in_proj_w = (const float*)d_in[1];
  const float* conv_w    = (const float*)d_in[2];
  const float* conv_b    = (const float*)d_in[3];
  const float* dt_bias   = (const float*)d_in[4];
  const float* A_log     = (const float*)d_in[5];
  const float* Dp        = (const float*)d_in[6];
  const float* norm_w    = (const float*)d_in[7];
  const float* out_proj_w= (const float*)d_in[8];

  char* ws = (char*)d_ws;
  h16*   zx   = (h16*)  (ws + OFF_ZX);
  h16*   X16  = (h16*)  (ws + OFF_X16);
  h16*   W16  = (h16*)  (ws + OFF_W16);
  h16*   WO16 = (h16*)  (ws + OFF_WO16);
  h16*   xBCc = (h16*)  (ws + OFF_XBC);
  h16*   xdt  = (h16*)  (ws + OFF_XDT);
  float* dtB  = (float*)(ws + OFF_DT);
  float* aB   = (float*)(ws + OFF_A);
  float* acum = (float*)(ws + OFF_ACUM);
  float* csum = (float*)(ws + OFF_CSUM);
  float* Sc   = (float*)(ws + OFF_SC);
  h16*   Sin  = (h16*)  (ws + OFF_SIN);
  float* yBuf = (float*)(ws + OFF_Y);
  h16*   yn   = (h16*)  (ws + OFF_YN);

  // 1) fp32 -> f16 operands
  cvt_f32_f16<<<(MTOT * DM + 255) / 256, 256, 0, stream>>>(x_seq, X16, MTOT * DM);
  cvt_f32_f16<<<(DPROJ * DM + 255) / 256, 256, 0, stream>>>(in_proj_w, W16, DPROJ * DM);
  cvt_f32_f16<<<(DM * DI + 255) / 256, 256, 0, stream>>>(out_proj_w, WO16, DM * DI);

  // 2) in_proj GEMM: zx[MTOT][DPROJ] (f16 out)
  gemm_nt_f16<h16><<<dim3(MTOT / 128, (DPROJ + 127) / 128), 256, 0, stream>>>(
      X16, W16, zx, MTOT, DPROJ, DM, DPROJ);

  // 3) dt / a ; conv + silu + x*dt
  dt_kernel<<<(MTOT * NH) / 256, 256, 0, stream>>>(zx, dt_bias, A_log, dtB, aB);
  conv_kernel<<<((long)MTOT * CONVD + 255) / 256, 256, 0, stream>>>(
      zx, conv_w, conv_b, dtB, xBCc, xdt);

  // 4) per-chunk cumsum of a
  cumsum_kernel<<<BSZ * NH * NC, 256, 0, stream>>>(aB, acum, csum);

  // 5) per-chunk states (WMMA)
  chunk_states_kernel<<<BSZ * NC * NH, 256, 0, stream>>>(xdt, xBCc, acum, Sc);

  // 6) inter-chunk state scan
  state_scan_kernel<<<(BSZ * NH * HD * DSTATE) / 256, 256, 0, stream>>>(Sc, csum, Sin);

  // 7) chunk Y (WMMA attention-like)
  chunk_y_kernel<<<BSZ * NC * NH, 256, 0, stream>>>(xBCc, xdt, Sin, acum, Dp, yBuf);

  // 8) gate + RMSnorm
  gate_norm_kernel<<<MTOT, 256, 0, stream>>>(yBuf, zx, norm_w, yn);

  // 9) out_proj GEMM -> d_out (fp32)
  gemm_nt_f16<float><<<dim3(MTOT / 128, DM / 128), 256, 0, stream>>>(
      yn, WO16, (float*)d_out, MTOT, DM, DI, DM);
}
